// AttentionHead_6201932775612
// MI455X (gfx1250) — matrix operand order repaired and checked by reading, not compiled, not run on hardware
//
#include <hip/hip_runtime.h>

// Problem sizes (match reference)
constexpr int NB = 8;     // batch
constexpr int NT = 2048;  // sequence length
constexpr int ND = 1024;  // d_model
constexpr int NH = 64;    // d_head

typedef __attribute__((ext_vector_type(16))) __bf16 v16bf;
typedef __attribute__((ext_vector_type(8)))  __bf16 v8bf;
typedef __attribute__((ext_vector_type(8)))  float  v8f;

static __device__ inline v16bf join8(v8bf lo, v8bf hi) {
  v16bf r;
#pragma unroll
  for (int i = 0; i < 8; ++i) { r[i] = lo[i]; r[i + 8] = hi[i]; }
  return r;
}

static __device__ inline v8f wmma_bf16(v16bf a, v16bf b, v8f c) {
  // D(f32 16x16) = A(bf16 16x32) * B(bf16 32x16) + C
  return __builtin_amdgcn_wmma_f32_16x16x32_bf16(false, a, false, b,
                                                 (short)0, c, false, false);
}

// Async copy 16B/lane from global (base+off) into LDS (per-lane byte address).
// GVS mode: mem_addr = SADDR + VADDR32; VDST = LDS byte address.
static __device__ inline void async_b128_to_lds(unsigned lds_addr,
                                                unsigned goff,
                                                const void* gbase) {
  asm volatile("global_load_async_to_lds_b128 %0, %1, %2"
               :
               : "v"(lds_addr), "v"(goff),
                 "s"((unsigned long long)(uintptr_t)gbase)
               : "memory");
}

static __device__ inline void wait_async0() {
  asm volatile("s_wait_asynccnt 0x0" ::: "memory");
}

// ---------------------------------------------------------------------------
// Kernel 1: convert W [ND][NH] f32 -> WT [NH][ND] bf16 (B-operand friendly)
// ---------------------------------------------------------------------------
__global__ void wtrans_kernel(const float* __restrict__ Wq,
                              const float* __restrict__ Wk,
                              const float* __restrict__ Wv,
                              __bf16* __restrict__ wqT,
                              __bf16* __restrict__ wkT,
                              __bf16* __restrict__ wvT) {
  int idx = blockIdx.x * blockDim.x + threadIdx.x;
  if (idx >= ND * NH) return;
  int kd = idx / NH;  // d_model index
  int n  = idx % NH;  // head index
  wqT[n * ND + kd] = (__bf16)Wq[idx];
  wkT[n * ND + kd] = (__bf16)Wk[idx];
  wvT[n * ND + kd] = (__bf16)Wv[idx];
}

// ---------------------------------------------------------------------------
// Kernel 2: fused Q/K/V projection with double-buffered async W staging.
// Per block/K-step the 12 KB W slice is async-copied once into LDS
// (global_load_async_to_lds_b128, ASYNCcnt) and consumed by all 4 waves as
// low-latency ds_load_b128 B-operands for the 12-WMMA burst, while the next
// slice streams in. x is read once and converted in registers (A operand).
// Q,K stored row-major [B*T][NH] bf16; V stored transposed [B][NH][T] bf16.
// ---------------------------------------------------------------------------
__global__ __launch_bounds__(128)
void proj_kernel(const float* __restrict__ x,
                 const __bf16* __restrict__ wqT,
                 const __bf16* __restrict__ wkT,
                 const __bf16* __restrict__ wvT,
                 __bf16* __restrict__ qb,
                 __bf16* __restrict__ kb,
                 __bf16* __restrict__ vtb) {
  __shared__ __bf16 wlds[2][3][NH][32];  // double-buffered W slice: 24 KB

  const int wave   = threadIdx.x >> 5;
  const int lane   = threadIdx.x & 31;
  const int laneLo = lane & 15;
  const int hi     = lane >> 4;           // half-wave selector
  const int t0     = (blockIdx.x * 4 + wave) * 16;  // flat row tile over B*T

  const float* xrow = x + (size_t)(t0 + laneLo) * ND;

  // Staging split: 128 threads x (2 rows x 3 matrices) x 16B = 12 KB / step
  const int sn0 = threadIdx.x >> 2;  // row n base: 0..31
  const int sc  = threadIdx.x & 3;   // 16-byte chunk within 64B row

  auto stage = [&](int sbuf, int k) {
#pragma unroll
    for (int i = 0; i < 2; ++i) {
      const int n = sn0 + i * 32;
      const unsigned goff = (unsigned)(((n * ND) + k + sc * 8) * 2);  // bytes
      async_b128_to_lds((unsigned)(uintptr_t)&wlds[sbuf][0][n][sc * 8], goff, wqT);
      async_b128_to_lds((unsigned)(uintptr_t)&wlds[sbuf][1][n][sc * 8], goff, wkT);
      async_b128_to_lds((unsigned)(uintptr_t)&wlds[sbuf][2][n][sc * 8], goff, wvT);
    }
  };

  const v8f vzero = {0.f, 0.f, 0.f, 0.f, 0.f, 0.f, 0.f, 0.f};
  v8f accQ[4], accK[4], accV[4];
#pragma unroll
  for (int i = 0; i < 4; ++i) { accQ[i] = vzero; accK[i] = vzero; accV[i] = vzero; }

  // Prologue: stage first slice
  stage(0, 0);
  wait_async0();
  __syncthreads();

  int buf = 0;
  for (int k = 0; k < ND; k += 32) {
    // Kick off next slice; its latency hides behind this step's compute.
    if (k + 32 < ND) stage(buf ^ 1, k + 32);

    // A-tile (16x32 bf16): lane row = laneLo; K = 16*hi + 0..15 (contiguous)
    v8f f0 = *(const v8f*)(xrow + k + hi * 16);
    v8f f1 = *(const v8f*)(xrow + k + hi * 16 + 8);
    v16bf a;
#pragma unroll
    for (int i = 0; i < 8; ++i) { a[i] = (__bf16)f0[i]; a[i + 8] = (__bf16)f1[i]; }

    // 12-WMMA burst; B operands are ds_load_b128 from the staged slice.
#pragma unroll
    for (int nt = 0; nt < 4; ++nt) {
      const int n = nt * 16 + laneLo;
      accQ[nt] = wmma_bf16(a, *(const v16bf*)&wlds[buf][0][n][hi * 16], accQ[nt]);
      accK[nt] = wmma_bf16(a, *(const v16bf*)&wlds[buf][1][n][hi * 16], accK[nt]);
      accV[nt] = wmma_bf16(a, *(const v16bf*)&wlds[buf][2][n][hi * 16], accV[nt]);
    }

    // Next slice complete + all waves done with current before swap.
    wait_async0();
    __syncthreads();
    buf ^= 1;
  }

  // C/D layout: lane(0-15) vgpr r -> (M=r, N=laneLo); lanes 16-31 -> M=8+r
#pragma unroll
  for (int nt = 0; nt < 4; ++nt) {
    const int n = nt * 16 + laneLo;
#pragma unroll
    for (int r = 0; r < 8; ++r) {
      const int row = t0 + hi * 8 + r;
      qb[(size_t)row * NH + n] = (__bf16)accQ[nt][r];
      kb[(size_t)row * NH + n] = (__bf16)accK[nt][r];
      const int bb = row >> 11;            // T == 2048
      const int tt = row & (NT - 1);
      vtb[((size_t)bb * NH + n) * NT + tt] = (__bf16)accV[nt][r];
    }
  }
}

// ---------------------------------------------------------------------------
// Kernel 3: causal flash attention. One wave per 16-query tile.
// Per 32-key step: all K/V operand loads issued up front (V latency hidden
// behind score WMMAs + softmax), then 4 WMMA (scores) + softmax + 4 WMMA (PV).
// ---------------------------------------------------------------------------
__global__ __launch_bounds__(128)
void attn_kernel(const __bf16* __restrict__ qb,
                 const __bf16* __restrict__ kb,
                 const __bf16* __restrict__ vtb,
                 float* __restrict__ out) {
  __shared__ __bf16 pbuf[4][16][32];  // per-wave P tile (C-layout -> A-layout bounce)

  const int wave   = threadIdx.x >> 5;
  const int lane   = threadIdx.x & 31;
  const int laneLo = lane & 15;
  const int hi     = lane >> 4;

  const int tileFlat = (blockIdx.x * 4 + wave) * 16;  // flat over B*T
  const int bb = tileFlat >> 11;
  const int q0 = tileFlat & (NT - 1);

  // Q as two 16x32 bf16 A-tiles (head dims 0..31 and 32..63), kept in VGPRs
  const __bf16* qrow = qb + ((size_t)bb * NT + q0 + laneLo) * NH;
  const v16bf aQ0 = join8(*(const v8bf*)(qrow + hi * 16),
                          *(const v8bf*)(qrow + hi * 16 + 8));
  const v16bf aQ1 = join8(*(const v8bf*)(qrow + 32 + hi * 16),
                          *(const v8bf*)(qrow + 40 + hi * 16));

  const v8f vzero = {0.f, 0.f, 0.f, 0.f, 0.f, 0.f, 0.f, 0.f};
  v8f accO[4];
#pragma unroll
  for (int i = 0; i < 4; ++i) accO[i] = vzero;

  float m[8], l[8];
#pragma unroll
  for (int r = 0; r < 8; ++r) { m[r] = -3.0e38f; l[r] = 0.f; }

  const float scale = 0.03125f;  // C^{-1/2} = 1/sqrt(1024) (faithful quirk)
  const int kmax = q0 + 15;      // highest key any query in this tile can see

  for (int k0 = 0; k0 <= kmax; k0 += 32) {
    // --- issue ALL operand loads for this step up front ---
    v16bf kB0[2], kB1[2], vB[4];
#pragma unroll
    for (int j = 0; j < 2; ++j) {
      const __bf16* krow = kb + ((size_t)bb * NT + k0 + j * 16 + laneLo) * NH;
      kB0[j] = *(const v16bf*)(krow + hi * 16);        // head dims  0..31
      kB1[j] = *(const v16bf*)(krow + 32 + hi * 16);   // head dims 32..63
    }
#pragma unroll
    for (int nt = 0; nt < 4; ++nt) {
      const __bf16* vcol =
          vtb + ((size_t)bb * NH + nt * 16 + laneLo) * NT + k0 + hi * 16;
      vB[nt] = *(const v16bf*)vcol;   // independent of scores/softmax
    }

    // --- scores: two 16-key N-tiles, K = 64 head dims (2 WMMA each) ---
    v8f s[2];
#pragma unroll
    for (int j = 0; j < 2; ++j) {
      v8f z = vzero;
      z = wmma_bf16(aQ0, kB0[j], z);
      z = wmma_bf16(aQ1, kB1[j], z);
      s[j] = z;
    }

    // --- online softmax per row (row M lives across 16 lanes of one half) ---
#pragma unroll
    for (int r = 0; r < 8; ++r) {
      const int qidx = q0 + hi * 8 + r;
      float s0 = s[0][r] * scale;
      float s1 = s[1][r] * scale;
      if (k0 + laneLo > qidx)      s0 = -3.0e38f;  // causal mask
      if (k0 + 16 + laneLo > qidx) s1 = -3.0e38f;

      float rmax = fmaxf(s0, s1);
      rmax = fmaxf(rmax, __shfl_xor(rmax, 1, 32));
      rmax = fmaxf(rmax, __shfl_xor(rmax, 2, 32));
      rmax = fmaxf(rmax, __shfl_xor(rmax, 4, 32));
      rmax = fmaxf(rmax, __shfl_xor(rmax, 8, 32));

      const float mnew  = fmaxf(m[r], rmax);
      const float alpha = __expf(m[r] - mnew);
      const float p0 = __expf(s0 - mnew);
      const float p1 = __expf(s1 - mnew);

      float rsum = p0 + p1;
      rsum += __shfl_xor(rsum, 1, 32);
      rsum += __shfl_xor(rsum, 2, 32);
      rsum += __shfl_xor(rsum, 4, 32);
      rsum += __shfl_xor(rsum, 8, 32);

      l[r] = l[r] * alpha + rsum;
      m[r] = mnew;
#pragma unroll
      for (int nt = 0; nt < 4; ++nt) accO[nt][r] *= alpha;

      pbuf[wave][hi * 8 + r][laneLo]      = (__bf16)p0;
      pbuf[wave][hi * 8 + r][16 + laneLo] = (__bf16)p1;
    }

    // --- re-stage P (16x32) from C-layout to A-layout via LDS ---
    const __bf16* prow = &pbuf[wave][laneLo][0];
    const v16bf aP = join8(*(const v8bf*)(prow + hi * 16),
                           *(const v8bf*)(prow + hi * 16 + 8));

    // --- PV: 4 head-dim N-tiles, K = 32 keys each (operands already loaded) ---
#pragma unroll
    for (int nt = 0; nt < 4; ++nt)
      accO[nt] = wmma_bf16(aP, vB[nt], accO[nt]);
  }

  // --- normalize and write fp32 output [B][T][NH] ---
#pragma unroll
  for (int nt = 0; nt < 4; ++nt) {
    const int n = nt * 16 + laneLo;
#pragma unroll
    for (int r = 0; r < 8; ++r) {
      const int row = tileFlat + hi * 8 + r;
      out[(size_t)row * NH + n] = accO[nt][r] / l[r];
    }
  }
}

// ---------------------------------------------------------------------------
extern "C" void kernel_launch(void* const* d_in, const int* in_sizes, int n_in,
                              void* d_out, int out_size, void* d_ws, size_t ws_size,
                              hipStream_t stream) {
  const float* x  = (const float*)d_in[0];
  const float* Wq = (const float*)d_in[1];
  const float* Wk = (const float*)d_in[2];
  const float* Wv = (const float*)d_in[3];
  float* out = (float*)d_out;

  // Workspace layout (bf16): q, k, vT, WqT, WkT, WvT  (~6.7 MB total)
  __bf16* qb  = (__bf16*)d_ws;
  __bf16* kb  = qb  + (size_t)NB * NT * NH;
  __bf16* vtb = kb  + (size_t)NB * NT * NH;
  __bf16* wqT = vtb + (size_t)NB * NH * NT;
  __bf16* wkT = wqT + (size_t)NH * ND;
  __bf16* wvT = wkT + (size_t)NH * ND;

  wtrans_kernel<<<(ND * NH + 255) / 256, 256, 0, stream>>>(Wq, Wk, Wv, wqT, wkT, wvT);

  // 16 rows per wave, 4 waves per block
  proj_kernel<<<(NB * NT) / 64, 128, 0, stream>>>(x, wqT, wkT, wvT, qb, kb, vtb);

  // 16 queries per wave, 4 waves per block
  attn_kernel<<<(NB * NT) / 64, 128, 0, stream>>>(qb, kb, vtb, out);
}
